// DGCNb_36644660969476
// MI455X (gfx1250) — matrix operand order repaired and checked by reading, not compiled, not run on hardware
//
#include <hip/hip_runtime.h>
#include <hip/hip_bf16.h>

typedef __attribute__((ext_vector_type(16))) __bf16 v16bf;
typedef __attribute__((ext_vector_type(8)))  float  v8f;

#define N_NODES 8192
#define NIN_DIM 2000
#define NZ_DIM  10
#define NC_DIM  10
#define EPSV    1e-10f

#define BM 128
#define BN 128
#define BK 32

#if __has_builtin(__builtin_amdgcn_cvt_pk_bf16_f32)
typedef __attribute__((ext_vector_type(2))) __bf16 v2bf;
__device__ __forceinline__ unsigned int pack_bf16(float lo, float hi) {
  union { v2bf v; unsigned int u; } c;
  c.v = __builtin_amdgcn_cvt_pk_bf16_f32(lo, hi);
  return c.u;
}
#else
// round-half-up bf16 pair pack: 2x v_add_nc_u32 + 1x v_perm_b32
__device__ __forceinline__ unsigned int pack_bf16(float lo, float hi) {
  unsigned int ulo = __float_as_uint(lo) + 0x8000u;
  unsigned int uhi = __float_as_uint(hi) + 0x8000u;
  return __builtin_amdgcn_perm(uhi, ulo, 0x07060302u);  // {uhi[31:16], ulo[31:16]}
}
#endif

union FragBF { uint4 q[2]; v16bf v; };

// C[M,N] = act( blend(A0,A1)[M,K] @ B[K,N] + bias ),  M == 8192 (multiple of BM)
// blend = alphaA * (reluA0 ? relu(A0) : A0) + betaA * A1   (A1 may be null)
// act: 0 = none, 1 = relu, 2 = clip(softplus(x), 0, 1e4)
__global__ __launch_bounds__(256) void gemm_bf16_wmma(
    const float* __restrict__ A0, const float* __restrict__ A1,
    float alphaA, float betaA, int reluA0,
    const float* __restrict__ B, const float* __restrict__ bias,
    float* __restrict__ C, int N, int K, int act)
{
  __shared__ unsigned int As[BM][BK / 2];   // [row][kpair]  (bf16 x2 per uint)
  __shared__ unsigned int Bs[BN][BK / 2];   // transposed: [n][kpair]

  const int tid      = threadIdx.x;
  const int lane     = tid & 31;
  const int wave     = tid >> 5;
  const int wm       = wave >> 1;           // 0..3 : 32-row slab
  const int wn       = wave & 1;            // 0..1 : 64-col slab
  const int lane15   = lane & 15;
  const int laneHalf = lane >> 4;
  const long long blockM = (long long)blockIdx.y * BM;
  const int blockN = blockIdx.x * BN;

  const bool vecA  = (K % 4 == 0);          // chunks are 4-aligned: all-in or all-out
  const bool vecB  = (N % 4 == 0);
  const bool hasA1 = (A1 != nullptr);

  v8f acc[2][4];
  {
    v8f zero = {};
#pragma unroll
    for (int i = 0; i < 2; ++i)
#pragma unroll
      for (int j = 0; j < 4; ++j) acc[i][j] = zero;
  }

  // ---- A staging mapping: thread -> (row = tid>>1, 16 cols at (tid&1)*16) ----
  const int arow = tid >> 1;
  const int acb  = (tid & 1) * 16;
  const float* a0row = A0 + (blockM + arow) * (long long)K + acb;
  const float* a1row = hasA1 ? (A1 + (blockM + arow) * (long long)K + acb) : nullptr;
  // ---- B staging mapping: thread -> (k-pair = tid>>4, 8 cols at (tid&15)*8) ----
  const int bkp = tid >> 4;                 // 0..15 : covers k = 2*bkp, 2*bkp+1
  const int bnb = (tid & 15) * 8;           // 0..120

  float aR[16];
  float bR[2][8];

  auto loadA = [&](int k0) {
#pragma unroll
    for (int c = 0; c < 4; ++c) {
      float4 v0 = {0.f, 0.f, 0.f, 0.f}, v1 = {0.f, 0.f, 0.f, 0.f};
      int kk = k0 + acb + 4 * c;
      if (vecA) {
        if (kk < K) {                        // K%4==0 -> whole chunk in-bounds
          v0 = *(const float4*)(a0row + k0 + 4 * c);
          if (hasA1) v1 = *(const float4*)(a1row + k0 + 4 * c);
        }
      } else {
#pragma unroll
        for (int e = 0; e < 4; ++e) {
          if (kk + e < K) {
            (&v0.x)[e] = a0row[k0 + 4 * c + e];
            if (hasA1) (&v1.x)[e] = a1row[k0 + 4 * c + e];
          }
        }
      }
#pragma unroll
      for (int e = 0; e < 4; ++e) {
        float x0 = (&v0.x)[e];
        if (reluA0) x0 = fmaxf(x0, 0.f);
        float r = alphaA * x0;
        if (hasA1) r += betaA * (&v1.x)[e];
        aR[4 * c + e] = r;
      }
    }
  };

  auto loadB = [&](int k0) {
#pragma unroll
    for (int r = 0; r < 2; ++r) {
      int gk = k0 + 2 * bkp + r;
      const float* bp = B + (long long)gk * N + blockN + bnb;
      bool rowOk = (gk < K);
#pragma unroll
      for (int c = 0; c < 2; ++c) {
        float4 v = {0.f, 0.f, 0.f, 0.f};
        int gn = blockN + bnb + 4 * c;
        if (vecB) {
          if (rowOk && gn < N)               // N%4==0 -> whole chunk in-bounds
            v = *(const float4*)(bp + 4 * c);
        } else {
#pragma unroll
          for (int e = 0; e < 4; ++e)
            if (rowOk && (gn + e < N)) (&v.x)[e] = bp[4 * c + e];
        }
#pragma unroll
        for (int e = 0; e < 4; ++e) bR[r][4 * c + e] = (&v.x)[e];
      }
    }
  };

  auto storeLDS = [&]() {
    unsigned int ua[8];
#pragma unroll
    for (int c = 0; c < 8; ++c) ua[c] = pack_bf16(aR[2 * c], aR[2 * c + 1]);
    uint4* dst = (uint4*)&As[arow][acb >> 1];
    dst[0] = make_uint4(ua[0], ua[1], ua[2], ua[3]);
    dst[1] = make_uint4(ua[4], ua[5], ua[6], ua[7]);
#pragma unroll
    for (int c = 0; c < 8; ++c)
      Bs[bnb + c][bkp] = pack_bf16(bR[0][c], bR[1][c]);
  };

  loadA(0);
  loadB(0);
  for (int k0 = 0; k0 < K; k0 += BK) {
    storeLDS();
    __syncthreads();
    if (k0 + BK < K) {                       // prefetch next tile into registers
      loadA(k0 + BK);
      loadB(k0 + BK);
      if (k0 + 2 * BK < K) __builtin_prefetch(a0row + k0 + 2 * BK, 0, 0);
    }
    // ---- build WMMA fragments from LDS (128-bit ds reads) and MAC ----
    FragBF af[2];
#pragma unroll
    for (int mt = 0; mt < 2; ++mt) {
      const uint4* rp = (const uint4*)(&As[wm * 32 + mt * 16 + lane15][0]);
      af[mt].q[0] = rp[laneHalf];            // K = 8*laneHalf .. +7
      af[mt].q[1] = rp[2 + laneHalf];        // K = 16 + 8*laneHalf .. +7
    }
#pragma unroll
    for (int nt = 0; nt < 4; ++nt) {
      const uint4* rp = (const uint4*)(&Bs[wn * 64 + nt * 16 + lane15][0]);
      FragBF bf;
      bf.q[0] = rp[laneHalf];                // K = 8*laneHalf .. +7
      bf.q[1] = rp[2 + laneHalf];            // K = 16 + 8*laneHalf .. +7
#pragma unroll
      for (int mt = 0; mt < 2; ++mt)
        acc[mt][nt] = __builtin_amdgcn_wmma_f32_16x16x32_bf16(
            false, af[mt].v, false, bf.v, (short)0, acc[mt][nt], false, false);
    }
    __syncthreads();
  }

  // ---- epilogue: bias + activation + store (16x16 f32 C layout) ----
#pragma unroll
  for (int nt = 0; nt < 4; ++nt) {
    int col = blockN + wn * 64 + nt * 16 + lane15;
    if (col < N) {
      float bv = bias ? bias[col] : 0.f;
#pragma unroll
      for (int mt = 0; mt < 2; ++mt) {
#pragma unroll
        for (int v = 0; v < 8; ++v) {
          long long row = blockM + wm * 32 + mt * 16 + laneHalf * 8 + v;
          float r = acc[mt][nt][v] + bv;
          if (act == 1) {
            r = fmaxf(r, 0.f);
          } else if (act == 2) {
            float sp = (r > 20.f) ? r : log1pf(expf(r));
            r = fminf(sp, 1e4f);
          }
          C[row * N + col] = r;
        }
      }
    }
  }
}

__global__ __launch_bounds__(256) void softmax_rows(
    const float* __restrict__ in, float* __restrict__ out)
{
  int i = blockIdx.x * blockDim.x + threadIdx.x;
  if (i >= N_NODES) return;
  const float* r = in + (long long)i * NZ_DIM;
  float m = r[0];
#pragma unroll
  for (int j = 1; j < NZ_DIM; ++j) m = fmaxf(m, r[j]);
  float e[NZ_DIM];
  float s = 0.f;
#pragma unroll
  for (int j = 0; j < NZ_DIM; ++j) { e[j] = expf(r[j] - m); s += e[j]; }
  float inv = 1.f / s;
#pragma unroll
  for (int j = 0; j < NZ_DIM; ++j) out[(long long)i * NZ_DIM + j] = e[j] * inv;
}

// DEC Student-t soft assignment, V = 1 -> q = (1 + d2)^-1 normalized per row
__global__ __launch_bounds__(256) void dec_q(
    const float* __restrict__ h, const float* __restrict__ cl, float* __restrict__ q)
{
  int i = blockIdx.x * blockDim.x + threadIdx.x;
  if (i >= N_NODES) return;
  float hv[NZ_DIM];
#pragma unroll
  for (int k = 0; k < NZ_DIM; ++k) hv[k] = h[(long long)i * NZ_DIM + k];
  float qv[NC_DIM];
  float s = 0.f;
#pragma unroll
  for (int c = 0; c < NC_DIM; ++c) {
    float d2 = 0.f;
#pragma unroll
    for (int k = 0; k < NZ_DIM; ++k) {
      float d = hv[k] - cl[c * NZ_DIM + k];
      d2 += d * d;
    }
    float t = 1.f / (1.f + d2);
    qv[c] = t; s += t;
  }
  float inv = 1.f / s;
#pragma unroll
  for (int c = 0; c < NC_DIM; ++c) q[(long long)i * NC_DIM + c] = qv[c] * inv;
}

__global__ __launch_bounds__(256) void nb_loss_partial(
    const float* __restrict__ y, const float* __restrict__ mu,
    const float* __restrict__ th, float* __restrict__ partials)
{
  __shared__ float sm[256];
  const long long total = (long long)N_NODES * NIN_DIM;
  float s = 0.f;
  for (long long idx = (long long)blockIdx.x * blockDim.x + threadIdx.x; idx < total;
       idx += (long long)gridDim.x * blockDim.x) {
    float t  = th[idx];
    float yt = y[idx];
    float yp = mu[idx];
    float t1 = lgammaf(t + EPSV) + lgammaf(yt + 1.f) - lgammaf(yt + t + EPSV);
    float t2 = (t + yt) * log1pf(yp / (t + EPSV)) + yt * (logf(t + EPSV) - logf(yp + EPSV));
    float v = t1 + t2;
    if (v != v) v = __builtin_inff();   // nan -> inf
    s += v;
  }
  sm[threadIdx.x] = s;
  __syncthreads();
  for (int o = 128; o > 0; o >>= 1) {
    if ((int)threadIdx.x < o) sm[threadIdx.x] += sm[threadIdx.x + o];
    __syncthreads();
  }
  if (threadIdx.x == 0) partials[blockIdx.x] = sm[0];
}

__global__ __launch_bounds__(256) void nb_loss_final(
    const float* __restrict__ partials, int n, float scale,
    float* __restrict__ loss, float* __restrict__ adjbar)
{
  __shared__ float sm[256];
  float s = 0.f;
  for (int i = threadIdx.x; i < n; i += 256) s += partials[i];
  sm[threadIdx.x] = s;
  __syncthreads();
  for (int o = 128; o > 0; o >>= 1) {
    if ((int)threadIdx.x < o) sm[threadIdx.x] += sm[threadIdx.x + o];
    __syncthreads();
  }
  if (threadIdx.x == 0) { loss[0] = sm[0] * scale; adjbar[0] = 0.f; }
}

static inline dim3 gemm_grid(int n) { return dim3((unsigned)((n + BN - 1) / BN), N_NODES / BM); }

extern "C" void kernel_launch(void* const* d_in, const int* in_sizes, int n_in,
                              void* d_out, int out_size, void* d_ws, size_t ws_size,
                              hipStream_t stream) {
  (void)in_sizes; (void)n_in; (void)out_size; (void)ws_size;
  const float s = 0.3f, os = 0.7f;   // SIGMA, 1-SIGMA

  const float* x      = (const float*)d_in[0];
  const float* adj    = (const float*)d_in[1];
  const float* enc1_w = (const float*)d_in[2];
  const float* enc1_b = (const float*)d_in[3];
  const float* enc2_w = (const float*)d_in[4];
  const float* enc2_b = (const float*)d_in[5];
  const float* enc3_w = (const float*)d_in[6];
  const float* enc3_b = (const float*)d_in[7];
  const float* z_w    = (const float*)d_in[8];
  const float* z_b    = (const float*)d_in[9];
  // d_in[10..17]: AE decoder params -> dead code in reference outputs, skipped
  const float* gnn1_w = (const float*)d_in[18];
  const float* gnn2_w = (const float*)d_in[19];
  const float* gnn3_w = (const float*)d_in[20];
  const float* gnn4_w = (const float*)d_in[21];
  const float* gnn5_w = (const float*)d_in[22];
  const float* fc1_w  = (const float*)d_in[23];
  const float* fc1_b  = (const float*)d_in[24];
  const float* disp_w = (const float*)d_in[25];
  const float* disp_b = (const float*)d_in[26];
  const float* mu_w   = (const float*)d_in[27];
  const float* mu_b   = (const float*)d_in[28];
  const float* clus   = (const float*)d_in[29];

  float* out = (float*)d_out;
  float* o_output = out;                     // [8192,2000]
  float* o_q      = o_output + 16384000;     // [8192,10]
  float* o_pred   = o_q + 81920;             // [8192,10]
  float* o_z      = o_pred + 81920;          // [8192,10]
  float* o_adjbar = o_z + 81920;             // scalar
  float* o_loss   = o_adjbar + 1;            // scalar
  float* o_h      = o_loss + 1;              // [8192,10]
  float* o_tra1   = o_h + 81920;             // [8192,500]
  float* o_tra2   = o_tra1 + 4096000;        // [8192,500]
  float* o_tra3   = o_tra2 + 4096000;        // [8192,2000]

  float* ws  = (float*)d_ws;
  float* ws0 = ws;                 // [8192,2000]  g3 -> later disp
  float* ws1 = ws0 + 16384000;     // [8192,2000]  h3 -> later deco
  float* ws2 = ws1 + 16384000;     // [8192,500]   g1/g2
  float* ws3 = ws2 + 4096000;      // [8192,500]   h1/h2
  float* ws4 = ws3 + 4096000;      // [8192,10]    g4/g5
  float* ws5 = ws4 + 81920;        // [8192,10]    h5
  float* wsP = ws5 + 81920;        // [4096]       loss partials

  dim3 blk(256);
  // ---- AE encoder ----
  gemm_bf16_wmma<<<gemm_grid(500),  blk, 0, stream>>>(x,      nullptr, 1.f, 0.f, 0, enc1_w, enc1_b, o_tra1, 500,  2000, 1);
  gemm_bf16_wmma<<<gemm_grid(500),  blk, 0, stream>>>(o_tra1, nullptr, 1.f, 0.f, 0, enc2_w, enc2_b, o_tra2, 500,  500,  1);
  gemm_bf16_wmma<<<gemm_grid(2000), blk, 0, stream>>>(o_tra2, nullptr, 1.f, 0.f, 0, enc3_w, enc3_b, o_tra3, 2000, 500,  1);
  gemm_bf16_wmma<<<gemm_grid(10),   blk, 0, stream>>>(o_tra3, nullptr, 1.f, 0.f, 0, z_w,    z_b,    o_z,    10,   2000, 0);
  // ---- GNN chain: h_{l} = relu(adj @ (mix @ W)) ----
  gemm_bf16_wmma<<<gemm_grid(500),  blk, 0, stream>>>(x,      nullptr, 1.f, 0.f, 0, gnn1_w, nullptr, ws2, 500,  2000, 0); // g1
  gemm_bf16_wmma<<<gemm_grid(500),  blk, 0, stream>>>(adj,    nullptr, 1.f, 0.f, 0, ws2,    nullptr, ws3, 500,  8192, 1); // h1
  gemm_bf16_wmma<<<gemm_grid(500),  blk, 0, stream>>>(ws3,    o_tra1,  os,  s,   0, gnn2_w, nullptr, ws2, 500,  500,  0); // g2
  gemm_bf16_wmma<<<gemm_grid(500),  blk, 0, stream>>>(adj,    nullptr, 1.f, 0.f, 0, ws2,    nullptr, ws3, 500,  8192, 1); // h2
  gemm_bf16_wmma<<<gemm_grid(2000), blk, 0, stream>>>(ws3,    o_tra2,  os,  s,   0, gnn3_w, nullptr, ws0, 2000, 500,  0); // g3
  gemm_bf16_wmma<<<gemm_grid(2000), blk, 0, stream>>>(adj,    nullptr, 1.f, 0.f, 0, ws0,    nullptr, ws1, 2000, 8192, 1); // h3
  gemm_bf16_wmma<<<gemm_grid(10),   blk, 0, stream>>>(ws1,    o_tra3,  os,  s,   0, gnn4_w, nullptr, ws4, 10,   2000, 0); // g4
  gemm_bf16_wmma<<<gemm_grid(10),   blk, 0, stream>>>(adj,    nullptr, 1.f, 0.f, 0, ws4,    nullptr, o_h, 10,   8192, 0); // h (no act)
  gemm_bf16_wmma<<<gemm_grid(10),   blk, 0, stream>>>(o_h,    o_z,     os,  s,   1, gnn5_w, nullptr, ws4, 10,   10,   0); // g5 (relu on h)
  gemm_bf16_wmma<<<gemm_grid(10),   blk, 0, stream>>>(adj,    nullptr, 1.f, 0.f, 0, ws4,    nullptr, ws5, 10,   8192, 0); // h5
  softmax_rows<<<dim3(N_NODES / 256), blk, 0, stream>>>(ws5, o_pred);
  // ---- NB decoder heads ----
  gemm_bf16_wmma<<<gemm_grid(1000), blk, 0, stream>>>(o_h, nullptr, 1.f, 0.f, 1, fc1_w,  fc1_b,  ws1,      1000, 10,   1); // deco
  gemm_bf16_wmma<<<gemm_grid(2000), blk, 0, stream>>>(ws1, nullptr, 1.f, 0.f, 0, disp_w, disp_b, ws0,      2000, 1000, 2); // disp
  gemm_bf16_wmma<<<gemm_grid(2000), blk, 0, stream>>>(ws1, nullptr, 1.f, 0.f, 0, mu_w,   mu_b,   o_output, 2000, 1000, 1); // output
  // ---- DEC q + NB loss ----
  dec_q<<<dim3(N_NODES / 256), blk, 0, stream>>>(o_h, clus, o_q);
  nb_loss_partial<<<dim3(4096), blk, 0, stream>>>(x, o_output, ws0, wsP);
  nb_loss_final<<<dim3(1), blk, 0, stream>>>(wsP, 4096, 1.f / ((float)N_NODES * (float)NIN_DIM), o_loss, o_adjbar);
}
